// RQVAE_90933047591096
// MI455X (gfx1250) — compile-verified, matrix operand order checked by reading, not run on hardware
//
#include <hip/hip_runtime.h>

typedef __attribute__((ext_vector_type(16))) _Float16 v16h;
typedef __attribute__((ext_vector_type(8)))  _Float16 v8h;
typedef __attribute__((ext_vector_type(8)))  float    v8f;
typedef __attribute__((ext_vector_type(4)))  float    v4f;

#define N_ROWS   131072
#define WAVES_PB 8
#define ROWS_PB  (WAVES_PB * 16)

// ---- workspace layout (units: halfs) -------------------------------------
// Weights pre-swizzled into WMMA 16x16x32 B-fragment order:
// frag(kt,nt) = 512 halfs; within a frag offset = lane*16 + e, where
//   k = kt*32 + (e%8) + 8*(lane>=16) + 16*(e/8),  n = nt*16 + (lane%16).
#define OFF_W1   0         // enc_w1 [768,128] : 24x8  frags (8KB per kt-slice)
#define OFF_W2   98304     // enc_w2 [128, 32] :  4x2
#define OFF_QIN  102400    // qin_w  3x[32,64] :  1x4 per layer (2048 each)
#define OFF_QOUT 108544    // qout_w 3x[64,32] :  2x2 per layer (2048 each)
#define OFF_D1   114688    // dec_w1 [32,128]  :  1x8
#define OFF_D2   118784    // dec_w2 [128,768] :  4x48
#define OFF_EMB  217088    // emb^T  3x[64,512]:  2x32 per layer (32768 each)
#define TOT_HALF 315392    // cnorm (3*512 f32) follows at byte 2*TOT_HALF

static __device__ __forceinline__ v8f wmma16(v16h a, v16h b, v8f c) {
  return __builtin_amdgcn_wmma_f32_16x16x32_f16(false, a, false, b,
                                                (short)0, c, false, false);
}

// A-fragment (16x32 f16, M=lane%16) from a row of f32 data
static __device__ __forceinline__ v16h a_from_f32(const float* p, int hi) {
  const float* p0 = p + 8 * hi;        // elems 0..7  -> k = 8*hi + 0..7
  const float* p1 = p + 16 + 8 * hi;   // elems 8..15 -> k = 16 + 8*hi + 0..7
  v4f x0 = *(const v4f*)p0, x1 = *(const v4f*)(p0 + 4);
  v4f x2 = *(const v4f*)p1, x3 = *(const v4f*)(p1 + 4);
  v16h a;
#pragma unroll
  for (int i = 0; i < 4; ++i) {
    a[i]      = (_Float16)x0[i];
    a[4 + i]  = (_Float16)x1[i];
    a[8 + i]  = (_Float16)x2[i];
    a[12 + i] = (_Float16)x3[i];
  }
  return a;
}

// A-fragment from LDS f16 row (row-major)
static __device__ __forceinline__ v16h a_from_lds(const _Float16* row, int koff, int hi) {
  v8h lo = *(const v8h*)(row + koff + 8 * hi);
  v8h hv = *(const v8h*)(row + koff + 16 + 8 * hi);
  v16h a;
#pragma unroll
  for (int i = 0; i < 8; ++i) { a[i] = lo[i]; a[8 + i] = hv[i]; }
  return a;
}

// B fragment from an LDS pool (contiguous 32B per lane -> 2x ds_load_b128)
static __device__ __forceinline__ v16h b_lds(const _Float16* pool, int off, int lane) {
  return *(const v16h*)(pool + off + lane * 16);
}

// cooperative global(f16 frags) -> LDS copy, v16h granules, 256 threads
static __device__ __forceinline__ void coop_copy(_Float16* dst, const _Float16* src,
                                                 int halfs, int tid) {
  int units = halfs >> 4;
  for (int i = tid; i < units; i += 256)
    *((v16h*)dst + i) = *((const v16h*)src + i);
}

// ---- prep: pack f32 matrix (element (k,n) at src[k*ks + n*ns]) into frags
__global__ void rq_pack_frags(const float* __restrict__ src, _Float16* __restrict__ dst,
                              int KT, int NT, int ks, int ns) {
  int tot = KT * NT * 512;
  for (int t = blockIdx.x * blockDim.x + threadIdx.x; t < tot;
       t += gridDim.x * blockDim.x) {
    int f = t >> 9, r = t & 511;
    int lane = r >> 4, e = r & 15;
    int kt = f / NT, nt = f - kt * NT;
    int hi = lane >> 4, nin = lane & 15;
    int kin = (e & 7) + 8 * hi + 16 * (e >> 3);
    int k = kt * 32 + kin, n = nt * 16 + nin;
    dst[t] = (_Float16)src[(size_t)k * ks + (size_t)n * ns];
  }
}

__global__ void rq_cnorm(const float* __restrict__ emb, float* __restrict__ cnorm) {
  int i = blockIdx.x * blockDim.x + threadIdx.x;
  if (i < 3 * 512) {
    const float* p = emb + (size_t)i * 64;
    float s = 0.f;
#pragma unroll
    for (int j = 0; j < 64; ++j) s += p[j] * p[j];
    cnorm[i] = 0.5f * s;
  }
}

// ---- fully fused RQ-VAE forward: one 16-row tile per wave -------------------
__global__ __launch_bounds__(256)
void rqvae_fused(const float* __restrict__ x,
                 const float* __restrict__ enc_b1, const float* __restrict__ enc_b2,
                 const float* __restrict__ dec_b1, const float* __restrict__ dec_b2,
                 const float* __restrict__ emb,
                 const float* __restrict__ qin_b, const float* __restrict__ qout_b,
                 const _Float16* __restrict__ wf, const float* __restrict__ cnorm,
                 float* __restrict__ out_recon, float* __restrict__ out_z,
                 float* __restrict__ out_q) {
  __shared__ __attribute__((aligned(32))) _Float16 s_stage[WAVES_PB][16 * 128]; // 32KB
  __shared__ __attribute__((aligned(32))) _Float16 s_pool[36864];               // 72KB
  __shared__ float s_cn[512];                                                   // 2KB
  __shared__ int   s_idx[WAVES_PB][16];

  const int tid  = threadIdx.x;
  const int wave = tid >> 5;
  const int lane = tid & 31;
  const int nlo  = lane & 15;
  const int hi   = lane >> 4;
  const int m    = nlo;
  const int rowBase = (blockIdx.x * WAVES_PB + wave) * 16;

  _Float16* stage = &s_stage[wave][0];
  int*      idxA  = &s_idx[wave][0];

  // ===== encoder GEMM1: h = relu(x @ W1 + b1)  [16,128] =====
  // W1 k-slices (8 frags = 4096 halfs) double-buffered in LDS pool.
  v16h wtmp = *(const v16h*)(wf + OFF_W1 + tid * 16);   // slice 0 preload
  v8f h[8] = {};
  const float* xrow = x + (size_t)(rowBase + m) * 768;
  for (int kt = 0; kt < 24; ++kt) {
    _Float16* buf = s_pool + (kt & 1) * 4096;
    *(v16h*)(buf + tid * 16) = wtmp;
    __syncthreads();
    if (kt < 23)
      wtmp = *(const v16h*)(wf + OFF_W1 + (kt + 1) * 4096 + tid * 16);
    if (kt + 1 < 24) __builtin_prefetch(xrow + (kt + 1) * 32, 0, 1);
    v16h a = a_from_f32(xrow + kt * 32, hi);
#pragma unroll
    for (int nt = 0; nt < 8; ++nt)
      h[nt] = wmma16(a, b_lds(buf, nt * 512, lane), h[nt]);
  }
#pragma unroll
  for (int nt = 0; nt < 8; ++nt) {
    float bias = enc_b1[nt * 16 + nlo];
#pragma unroll
    for (int r = 0; r < 8; ++r) {
      float v = h[nt][r] + bias;
      stage[(r + 8 * hi) * 128 + nt * 16 + nlo] = (_Float16)(v > 0.f ? v : 0.f);
    }
  }

  // ===== encoder GEMM2: z = h @ W2 + b2  [16,32] =====
  __syncthreads();                         // enc1 pool readers done
  coop_copy(s_pool, wf + OFF_W2, 4096, tid);
  __syncthreads();
  v8f z[2] = {};
#pragma unroll
  for (int kt = 0; kt < 4; ++kt) {
    v16h a = a_from_lds(stage + m * 128, kt * 32, hi);
#pragma unroll
    for (int nt = 0; nt < 2; ++nt)
      z[nt] = wmma16(a, b_lds(s_pool, (kt * 2 + nt) * 512, lane), z[nt]);
  }
  v8f resid[2], quant[2];
#pragma unroll
  for (int nt = 0; nt < 2; ++nt) {
    float bias = enc_b2[nt * 16 + nlo];
#pragma unroll
    for (int r = 0; r < 8; ++r) {
      float v = z[nt][r] + bias;
      out_z[(size_t)(rowBase + r + 8 * hi) * 32 + nt * 16 + nlo] = v;
      resid[nt][r] = v;
      quant[nt][r] = 0.f;
    }
  }

  // ===== 3 residual-VQ layers =====
  for (int l = 0; l < 3; ++l) {
    __syncthreads();                       // previous pool readers done
    coop_copy(s_pool,         wf + OFF_EMB  + l * 32768, 32768, tid);  // EMB_l
    coop_copy(s_pool + 32768, wf + OFF_QIN  + l * 2048,  2048,  tid);  // QIN_l
    coop_copy(s_pool + 34816, wf + OFF_QOUT + l * 2048,  2048,  tid);  // QOUT_l
    for (int i = tid; i < 512; i += 256) s_cn[i] = cnorm[l * 512 + i];
    __syncthreads();

    // stage residual as f16 [16][32]
#pragma unroll
    for (int nt = 0; nt < 2; ++nt)
#pragma unroll
      for (int r = 0; r < 8; ++r)
        stage[(r + 8 * hi) * 32 + nt * 16 + nlo] = (_Float16)resid[nt][r];

    // f = residual @ qin_w[l] + qin_b[l]   [16,64]
    v8f f[4] = {};
    {
      v16h a = a_from_lds(stage + m * 32, 0, hi);
#pragma unroll
      for (int nt = 0; nt < 4; ++nt)
        f[nt] = wmma16(a, b_lds(s_pool, 32768 + nt * 512, lane), f[nt]);
    }
#pragma unroll
    for (int nt = 0; nt < 4; ++nt) {
      float bias = qin_b[l * 64 + nt * 16 + nlo];
#pragma unroll
      for (int r = 0; r < 8; ++r)
        stage[(r + 8 * hi) * 64 + nt * 16 + nlo] = (_Float16)(f[nt][r] + bias);
    }
    v16h fa0 = a_from_lds(stage + m * 64, 0, hi);
    v16h fa1 = a_from_lds(stage + m * 64, 32, hi);

    // argmin over 512 codes  <=>  argmax(f.c - 0.5*||c||^2)
    float best[8];
    int   bidx[8];
#pragma unroll
    for (int r = 0; r < 8; ++r) { best[r] = -3.0e38f; bidx[r] = 0; }
#pragma unroll 2
    for (int nt = 0; nt < 32; ++nt) {
      v8f s = {};
      s = wmma16(fa0, b_lds(s_pool, nt * 512, lane), s);
      s = wmma16(fa1, b_lds(s_pool, (32 + nt) * 512, lane), s);
      int   n  = nt * 16 + nlo;
      float cn = s_cn[n];
#pragma unroll
      for (int r = 0; r < 8; ++r) {
        float sc = s[r] - cn;
        if (sc > best[r]) { best[r] = sc; bidx[r] = n; }
      }
    }
    for (int off = 1; off < 16; off <<= 1) {
#pragma unroll
      for (int r = 0; r < 8; ++r) {
        float ob = __shfl_xor(best[r], off, 16);
        int   oi = __shfl_xor(bidx[r], off, 16);
        if (ob > best[r] || (ob == best[r] && oi < bidx[r])) {
          best[r] = ob; bidx[r] = oi;
        }
      }
    }
    if (nlo == 0) {
#pragma unroll
      for (int r = 0; r < 8; ++r) idxA[r + 8 * hi] = bidx[r];
    }
    int code = idxA[m];

    // q = emb[l][code] @ qout_w[l] + qout_b[l]; straight-through => value q
    const float* crow = emb + ((size_t)l * 512 + code) * 64;
    v16h ca0 = a_from_f32(crow, hi);
    v16h ca1 = a_from_f32(crow + 32, hi);
    v8f q[2] = {};
#pragma unroll
    for (int nt = 0; nt < 2; ++nt) {
      q[nt] = wmma16(ca0, b_lds(s_pool, 34816 + nt * 512, lane), q[nt]);
      q[nt] = wmma16(ca1, b_lds(s_pool, 34816 + (2 + nt) * 512, lane), q[nt]);
    }
#pragma unroll
    for (int nt = 0; nt < 2; ++nt) {
      float bias = qout_b[l * 32 + nt * 16 + nlo];
#pragma unroll
      for (int r = 0; r < 8; ++r) {
        float qq = q[nt][r] + bias;
        quant[nt][r] += qq;
        resid[nt][r] -= qq;
      }
    }
  }

  // write quantized output
#pragma unroll
  for (int nt = 0; nt < 2; ++nt)
#pragma unroll
    for (int r = 0; r < 8; ++r)
      out_q[(size_t)(rowBase + r + 8 * hi) * 32 + nt * 16 + nlo] = quant[nt][r];

  // ===== decoder: recon = relu(quant @ D1 + b1) @ D2 + b2 =====
  __syncthreads();                          // RVQ pool readers done
  coop_copy(s_pool, wf + OFF_D1, 4096, tid);
  __syncthreads();

#pragma unroll
  for (int nt = 0; nt < 2; ++nt)
#pragma unroll
    for (int r = 0; r < 8; ++r)
      stage[(r + 8 * hi) * 32 + nt * 16 + nlo] = (_Float16)quant[nt][r];

  v8f h2[8] = {};
  {
    v16h a = a_from_lds(stage + m * 32, 0, hi);
#pragma unroll
    for (int nt = 0; nt < 8; ++nt)
      h2[nt] = wmma16(a, b_lds(s_pool, nt * 512, lane), h2[nt]);
  }
#pragma unroll
  for (int nt = 0; nt < 8; ++nt) {
    float bias = dec_b1[nt * 16 + nlo];
#pragma unroll
    for (int r = 0; r < 8; ++r) {
      float v = h2[nt][r] + bias;
      stage[(r + 8 * hi) * 128 + nt * 16 + nlo] = (_Float16)(v > 0.f ? v : 0.f);
    }
  }
  v16h da[4];
#pragma unroll
  for (int kt = 0; kt < 4; ++kt) da[kt] = a_from_lds(stage + m * 128, kt * 32, hi);

  // D2 in double-buffered 32KB slices: slice g = frags (kt, g*8+j)
  v16h d2t[4];
#pragma unroll
  for (int kt = 0; kt < 4; ++kt)
    d2t[kt] = *(const v16h*)(wf + OFF_D2 + (kt * 48) * 512 + tid * 16);
  for (int g = 0; g < 6; ++g) {
    _Float16* buf = s_pool + 4096 + (g & 1) * 16384;
#pragma unroll
    for (int kt = 0; kt < 4; ++kt)
      *(v16h*)(buf + kt * 4096 + tid * 16) = d2t[kt];
    __syncthreads();
    if (g < 5) {
#pragma unroll
      for (int kt = 0; kt < 4; ++kt)
        d2t[kt] = *(const v16h*)(wf + OFF_D2 + (kt * 48 + (g + 1) * 8) * 512 + tid * 16);
    }
    v8f o[8] = {};
#pragma unroll
    for (int kt = 0; kt < 4; ++kt)
#pragma unroll
      for (int j = 0; j < 8; ++j)
        o[j] = wmma16(da[kt], b_lds(buf, kt * 4096 + j * 512, lane), o[j]);
#pragma unroll
    for (int j = 0; j < 8; ++j) {
      int n = (g * 8 + j) * 16 + nlo;
      float bias = dec_b2[n];
#pragma unroll
      for (int r = 0; r < 8; ++r)
        out_recon[(size_t)(rowBase + r + 8 * hi) * 768 + n] = o[j][r] + bias;
    }
  }
}

// ---------------------------------------------------------------------------
extern "C" void kernel_launch(void* const* d_in, const int* in_sizes, int n_in,
                              void* d_out, int out_size, void* d_ws, size_t ws_size,
                              hipStream_t stream) {
  const float* x      = (const float*)d_in[0];
  const float* enc_w1 = (const float*)d_in[1];
  const float* enc_b1 = (const float*)d_in[2];
  const float* enc_w2 = (const float*)d_in[3];
  const float* enc_b2 = (const float*)d_in[4];
  const float* dec_w1 = (const float*)d_in[5];
  const float* dec_b1 = (const float*)d_in[6];
  const float* dec_w2 = (const float*)d_in[7];
  const float* dec_b2 = (const float*)d_in[8];
  const float* emb    = (const float*)d_in[9];
  const float* qin_w  = (const float*)d_in[10];
  const float* qin_b  = (const float*)d_in[11];
  const float* qout_w = (const float*)d_in[12];
  const float* qout_b = (const float*)d_in[13];
  (void)in_sizes; (void)n_in; (void)out_size; (void)ws_size;

  _Float16* wsH   = (_Float16*)d_ws;
  float*    cnorm = (float*)((char*)d_ws + (size_t)TOT_HALF * 2);

  auto pack = [&](const float* src, int off, int KT, int NT, int ks, int ns) {
    int tot = KT * NT * 512;
    rq_pack_frags<<<(tot + 255) / 256, 256, 0, stream>>>(src, wsH + off, KT, NT, ks, ns);
  };
  pack(enc_w1, OFF_W1, 24, 8, 128, 1);
  pack(enc_w2, OFF_W2, 4, 2, 32, 1);
  pack(dec_w1, OFF_D1, 1, 8, 128, 1);
  pack(dec_w2, OFF_D2, 4, 48, 768, 1);
  for (int l = 0; l < 3; ++l) {
    pack(qin_w  + (size_t)l * 32 * 64, OFF_QIN  + l * 2048, 1, 4, 64, 1);
    pack(qout_w + (size_t)l * 64 * 32, OFF_QOUT + l * 2048, 2, 2, 32, 1);
    // B = code^T: element (k,n) = emb[l][n][k]  -> ks=1, ns=64
    pack(emb + (size_t)l * 512 * 64, OFF_EMB + l * 32768, 2, 32, 1, 64);
  }
  rq_cnorm<<<6, 256, 0, stream>>>(emb, cnorm);

  float* out_recon = (float*)d_out;
  float* out_z = out_recon + (size_t)N_ROWS * 768;
  float* out_q = out_z + (size_t)N_ROWS * 32;

  rqvae_fused<<<N_ROWS / ROWS_PB, WAVES_PB * 32, 0, stream>>>(
      x, enc_b1, enc_b2, dec_b1, dec_b2, emb, qin_b, qout_b,
      wsH, cnorm, out_recon, out_z, out_q);
}
